// NonParametricModel_27779848471270
// MI455X (gfx1250) — matrix operand order repaired and checked by reading, not compiled, not run on hardware
//
#include <hip/hip_runtime.h>
#include <hip/hip_bf16.h>

typedef __attribute__((ext_vector_type(16))) _Float16 v16h;
typedef __attribute__((ext_vector_type(2)))  _Float16 v2h;
typedef __attribute__((ext_vector_type(8)))  float    v8f;
typedef __attribute__((ext_vector_type(8)))  int      v8i;

#define NCELLS   512
#define NBATCH   2048
#define NGROUPS  (NBATCH * NCELLS / 16)   // 65536 groups of 16 samples
#define OUTTOT   (NBATCH * 3 * NCELLS)

// ---------------------------------------------------------------------------
// helpers
// ---------------------------------------------------------------------------
__device__ __forceinline__ int swz16(int x) {
  // swap lane halves of the wave32: group-of-32 swizzle, xor=0x10, and=0x1f
  return __builtin_amdgcn_ds_swizzle(x, 0x401F);
}

__device__ __forceinline__ int pkh(float lo, float hi) {
  // v_cvt_pk_f16_f32
  v2h p = { (_Float16)lo, (_Float16)hi };
  return __builtin_bit_cast(int, p);
}

__device__ __forceinline__ int relu_pk(float lo, float hi) {
  // packed f16 relu: cvt_pk + v_pk_max_f16 (relu commutes with the downcast)
  v2h p = { (_Float16)lo, (_Float16)hi };
  v2h z = {};
  p = __builtin_elementwise_max(p, z);
  return __builtin_bit_cast(int, p);
}

__device__ __forceinline__ v8f wmma16(v16h a, v16h b, v8f c) {
  return __builtin_amdgcn_wmma_f32_16x16x32_f16(
      /*neg_a=*/false, a, /*neg_b=*/false, b,
      /*c_mod=*/(short)0, c, /*reuse_a=*/false, /*reuse_b=*/false);
}

// Build A fragment = W^T tile (M = output feature, K = input feature).
// A fragment k order used here (contiguous, same as B): lane l holds row m=l%16; element e=(v,pos):
//   k = 32*kblk + 16*(l>>4) + 2*v + pos
// If bias != nullptr, slot k == Kin carries bias[m] (bias-in-K trick).
__device__ __forceinline__ v16h load_afrag(const float* __restrict__ w,
                                           const float* __restrict__ bias,
                                           int Kin, int Nout, int mtile,
                                           int kblk, int lane) {
  const int m  = mtile * 16 + (lane & 15);
  const int hi = (lane >> 4) & 1;
  v16h a;
#pragma unroll
  for (int e = 0; e < 16; ++e) {
    const int v = e >> 1, pos = e & 1;
    const int k = kblk * 32 + (hi << 4) + (v << 1) + pos;
    float val = 0.f;
    if (k < Kin)               val = w[k * Nout + m];
    else if (bias && k == Kin) val = bias[m];
    a[e] = (_Float16)val;
  }
  return a;
}

// Pack next-layer B fragment (K = 16*hi + e, N = lane%16) from two f32 D tiles
// (feature tiles 2j and 2j+1), relu'd, all in packed-f16 dwords:
//   hi=0 lanes: dwords 0..3 = own tile-a pairs, dwords 4..7 = partner tile-a
//   hi=1 lanes: dwords 0..3 = partner tile-b pairs, dwords 4..7 = own tile-b
// One ds_swizzle moves two f16 values at a time.
__device__ __forceinline__ v16h pack_b(const v8f& da, const v8f& db, bool hi1) {
  v8i bi;
#pragma unroll
  for (int v = 0; v < 4; ++v) {
    const int pa = relu_pk(da[2 * v], da[2 * v + 1]);
    const int pb = relu_pk(db[2 * v], db[2 * v + 1]);
    const int sa = swz16(pa);
    const int sb = swz16(pb);
    bi[v]     = hi1 ? sb : pa;
    bi[v + 4] = hi1 ? pb : sa;
  }
  return __builtin_bit_cast(v16h, bi);
}

// 16-feature layer output -> B frag; K=16 slot (hi=1, dword0 lo) = 1.0 (bias)
__device__ __forceinline__ v16h pack_b16(const v8f& d0, bool hi1) {
  v8i bi;
#pragma unroll
  for (int v = 0; v < 4; ++v) {
    const int pa = relu_pk(d0[2 * v], d0[2 * v + 1]);
    const int sa = swz16(pa);
    bi[v]     = hi1 ? (v == 0 ? 0x3c00 : 0) : pa;
    bi[v + 4] = hi1 ? 0 : sa;
  }
  return __builtin_bit_cast(v16h, bi);
}

// ---------------------------------------------------------------------------
// Kernel 1 (hot): fused divergence + MLP2 (WMMA) + field = p * div
// ---------------------------------------------------------------------------
__global__ void __launch_bounds__(256)
field_kernel(const float* __restrict__ x,
             const float* __restrict__ w0, const float* __restrict__ b0v,
             const float* __restrict__ w1, const float* __restrict__ b1v,
             const float* __restrict__ w2, const float* __restrict__ b2v,
             const float* __restrict__ w3, const float* __restrict__ b3v,
             const float* __restrict__ w4, const float* __restrict__ b4v,
             float* __restrict__ field, int nWaves) {
  const int lane = threadIdx.x & 31;
  const int hi   = lane >> 4;
  const bool hi1 = hi != 0;
  const int wid  = (blockIdx.x * blockDim.x + threadIdx.x) >> 5;

  // ---- register-resident weight fragments (one-time, amortized) ----
  v16h A0 = load_afrag(w0, b0v, 6, 16, 0, 0, lane);          // 6->16  (bias @k=6)
  v16h A1[4];
#pragma unroll
  for (int t = 0; t < 4; ++t)
    A1[t] = load_afrag(w1, b1v, 16, 64, t, 0, lane);         // 16->64 (bias @k=16)
  v16h A2[4][2];
#pragma unroll
  for (int t = 0; t < 4; ++t)
#pragma unroll
    for (int kb = 0; kb < 2; ++kb)
      A2[t][kb] = load_afrag(w2, nullptr, 64, 64, t, kb, lane); // 64->64
  v16h A3[2];
#pragma unroll
  for (int kb = 0; kb < 2; ++kb)
    A3[kb] = load_afrag(w3, nullptr, 64, 16, 0, kb, lane);   // 64->16
  // biases for layers 2,3 as C accumulator seeds (no spare K slot there)
  v8f C2[4], C3;
#pragma unroll
  for (int t = 0; t < 4; ++t)
#pragma unroll
    for (int r = 0; r < 8; ++r) C2[t][r] = b2v[16 * t + 8 * hi + r];
#pragma unroll
  for (int r = 0; r < 8; ++r) C3[r] = b3v[8 * hi + r];
  float W4[8];
#pragma unroll
  for (int r = 0; r < 8; ++r) W4[r] = w4[8 * hi + r];        // [16,1] weights
  const float B4 = b4v[0];
  const v8f cz = {};

  for (int g = wid; g < NGROUPS; g += nWaves) {              // uniform per wave
    const int b        = g >> 5;
    const int cellbase = (g & 31) << 4;
    const int n    = lane & 15;
    const int cell = cellbase + n;
    const int h = cell >> 6, w = (cell >> 3) & 7, l = cell & 7;
    const float* xb = x + (size_t)b * 1536;

    const float x0 = xb[cell], x1 = xb[512 + cell], x2 = xb[1024 + cell];
    // circular central-difference divergence, 1/(2*DX) = 1/16
    const int hp = (((h + 1) & 7) << 6) | (w << 3) | l;
    const int hm = (((h + 7) & 7) << 6) | (w << 3) | l;
    const int wp = (h << 6) | (((w + 1) & 7) << 3) | l;
    const int wm = (h << 6) | (((w + 7) & 7) << 3) | l;
    const int lp = (h << 6) | (w << 3) | ((l + 1) & 7);
    const int lm = (h << 6) | (w << 3) | ((l + 7) & 7);
    const float div = (xb[hp] - xb[hm] + xb[512 + wp] - xb[512 + wm] +
                       xb[1024 + lp] - xb[1024 + lm]) * 0.0625f;

    // layer-0 B fragment as packed dwords: K = 16*hi + e; rows k>=8 of A0 are
    // zero, so the identical fill is harmless on hi=1 lanes.
    v8i b0i = {};
    b0i[0] = pkh(-0.875f + 0.25f * (float)h, -0.875f + 0.25f * (float)w);
    b0i[1] = pkh(-0.875f + 0.25f * (float)l, x0);
    b0i[2] = pkh(x1, x2);
    b0i[3] = 0x3c00;                                         // bias slot k=6
    v16h bf = __builtin_bit_cast(v16h, b0i);

    v8f d0 = wmma16(A0, bf, cz);                             // 6->16
    v16h b1f = pack_b16(d0, hi1);
    v8f d1[4];
#pragma unroll
    for (int t = 0; t < 4; ++t) d1[t] = wmma16(A1[t], b1f, cz);   // 16->64
    v16h b2a = pack_b(d1[0], d1[1], hi1);
    v16h b2b = pack_b(d1[2], d1[3], hi1);
    v8f d2[4];
#pragma unroll
    for (int t = 0; t < 4; ++t)
      d2[t] = wmma16(A2[t][1], b2b, wmma16(A2[t][0], b2a, C2[t])); // 64->64
    v16h b3a = pack_b(d2[0], d2[1], hi1);
    v16h b3b = pack_b(d2[2], d2[3], hi1);
    v8f d3 = wmma16(A3[1], b3b, wmma16(A3[0], b3a, C3));     // 64->16

    // final 16->1: per-lane 8-FMA dot over features 8*hi..8*hi+7, halves merged
    float pp = 0.f;
#pragma unroll
    for (int r = 0; r < 8; ++r)
      pp += W4[r] * __builtin_amdgcn_fmed3f(d3[r], 0.f, 3.0e38f); // relu, 1 op
    const float psum = pp +
        __int_as_float(swz16(__float_as_int(pp))) + B4;

    if (lane < 16) field[(size_t)b * 512 + cell] = psum * div;
  }
}

// ---------------------------------------------------------------------------
// Kernel 0 (tiny): z1 = MLP1(coords), batch-independent -> [3][512] scratch
// ---------------------------------------------------------------------------
__global__ void mlp1_kernel(const float* __restrict__ w0, const float* __restrict__ b0,
                            const float* __restrict__ w1, const float* __restrict__ b1,
                            const float* __restrict__ w2, const float* __restrict__ b2,
                            const float* __restrict__ w3, const float* __restrict__ b3,
                            const float* __restrict__ w4, const float* __restrict__ b4,
                            float* __restrict__ z1) {
  const int cell = blockIdx.x * blockDim.x + threadIdx.x;
  if (cell >= NCELLS) return;
  const int h = cell >> 6, w = (cell >> 3) & 7, l = cell & 7;
  const float in0 = -0.875f + 0.25f * (float)h;
  const float in1 = -0.875f + 0.25f * (float)w;
  const float in2 = -0.875f + 0.25f * (float)l;

  float h0[16];
#pragma unroll
  for (int j = 0; j < 16; ++j)
    h0[j] = fmaxf(b0[j] + in0 * w0[j] + in1 * w0[16 + j] + in2 * w0[32 + j], 0.f);
  float h1[64];
#pragma unroll
  for (int j = 0; j < 64; ++j) {
    float a = b1[j];
#pragma unroll
    for (int k = 0; k < 16; ++k) a += h0[k] * w1[k * 64 + j];
    h1[j] = fmaxf(a, 0.f);
  }
  float h2[64];
#pragma unroll
  for (int j = 0; j < 64; ++j) {
    float a = b2[j];
#pragma unroll
    for (int k = 0; k < 64; ++k) a += h1[k] * w2[k * 64 + j];
    h2[j] = fmaxf(a, 0.f);
  }
  float h3[16];
#pragma unroll
  for (int j = 0; j < 16; ++j) {
    float a = b3[j];
#pragma unroll
    for (int k = 0; k < 64; ++k) a += h2[k] * w3[k * 16 + j];
    h3[j] = fmaxf(a, 0.f);
  }
#pragma unroll
  for (int c = 0; c < 3; ++c) {
    float a = b4[c];
#pragma unroll
    for (int k = 0; k < 16; ++k) a += h3[k] * w4[k * 3 + c];
    z1[c * NCELLS + cell] = a;
  }
}

// ---------------------------------------------------------------------------
// Kernel 2: out = z1 + grad(field)   (memory-bound stencil)
// ---------------------------------------------------------------------------
__global__ void grad_kernel(const float* __restrict__ field,
                            const float* __restrict__ z1,
                            float* __restrict__ out) {
  const int idx = blockIdx.x * blockDim.x + threadIdx.x;
  if (idx >= OUTTOT) return;
  const int cell = idx & 511;
  const int c    = (idx >> 9) % 3;
  const int b    = idx / 1536;
  const int h = cell >> 6, w = (cell >> 3) & 7, l = cell & 7;
  int cp, cm;
  if (c == 0)      { cp = (((h + 1) & 7) << 6) | (w << 3) | l;
                     cm = (((h + 7) & 7) << 6) | (w << 3) | l; }
  else if (c == 1) { cp = (h << 6) | (((w + 1) & 7) << 3) | l;
                     cm = (h << 6) | (((w + 7) & 7) << 3) | l; }
  else             { cp = (h << 6) | (w << 3) | ((l + 1) & 7);
                     cm = (h << 6) | (w << 3) | ((l + 7) & 7); }
  const float* fb = field + (size_t)b * 512;
  out[idx] = z1[c * NCELLS + cell] + (fb[cp] - fb[cm]) * 0.0625f;
}

// ---------------------------------------------------------------------------
extern "C" void kernel_launch(void* const* d_in, const int* in_sizes, int n_in,
                              void* d_out, int out_size, void* d_ws, size_t ws_size,
                              hipStream_t stream) {
  const float* x = (const float*)d_in[0];
  const float *w1_[5], *b1_[5], *w2_[5], *b2_[5];
  for (int i = 0; i < 5; ++i) {
    w1_[i] = (const float*)d_in[1 + 2 * i];
    b1_[i] = (const float*)d_in[2 + 2 * i];
    w2_[i] = (const float*)d_in[11 + 2 * i];
    b2_[i] = (const float*)d_in[12 + 2 * i];
  }
  float* field = (float*)d_ws;                                   // 4 MB
  float* z1    = (float*)((char*)d_ws + (size_t)NBATCH * NCELLS * 4); // 6 KB

  mlp1_kernel<<<2, 256, 0, stream>>>(w1_[0], b1_[0], w1_[1], b1_[1], w1_[2],
                                     b1_[2], w1_[3], b1_[3], w1_[4], b1_[4], z1);

  const int blocks = 512, tpb = 256;
  const int nWaves = blocks * tpb / 32;
  field_kernel<<<blocks, tpb, 0, stream>>>(x,
      w2_[0], b2_[0], w2_[1], b2_[1], w2_[2], b2_[2], w2_[3], b2_[3],
      w2_[4], b2_[4], field, nWaves);

  grad_kernel<<<(OUTTOT + 255) / 256, 256, 0, stream>>>(field, z1, (float*)d_out);
}
